// MultiHeadSelfAttention_84224308675034
// MI455X (gfx1250) — matrix operand order repaired and checked by reading, not compiled, not run on hardware
//
#include <hip/hip_runtime.h>
#include <hip/hip_bf16.h>
#include <math.h>

#define B_  2
#define S_  2048
#define D_  1024
#define H_  16
#define DK_ 64

typedef _Float16 v16h __attribute__((ext_vector_type(16)));
typedef _Float16 v8h  __attribute__((ext_vector_type(8)));
typedef float    v8f  __attribute__((ext_vector_type(8)));

__device__ __forceinline__ v8f wmma_ff16(v16h a, v16h b, v8f c) {
  // D(16x16,f32) = A(16x32,f16) * B(32x16,f16) + C
  return __builtin_amdgcn_wmma_f32_16x16x32_f16(false, a, false, b, (short)0, c,
                                                false, false);
}

// LDS byte offset of a __shared__ object (AS(3) pointer value == LDS offset).
__device__ __forceinline__ unsigned lds_addr(const void* p) {
  return (unsigned)(unsigned long long)(const __attribute__((address_space(3))) char*)p;
}

// Async global->LDS tile staging (CDNA5; tracked by ASYNCcnt).
#define ASYNC_B128(ldsoff, gptr)                                        \
  asm volatile("global_load_async_to_lds_b128 %0, %1, off"              \
               ::"v"(ldsoff), "v"(gptr) : "memory")
#define ASYNC_B128_OFF16(ldsoff, gptr)                                  \
  asm volatile("global_load_async_to_lds_b128 %0, %1, off offset:16"    \
               ::"v"(ldsoff), "v"(gptr) : "memory")
#define ASYNC_B128_OFF32(ldsoff, gptr)                                  \
  asm volatile("global_load_async_to_lds_b128 %0, %1, off offset:32"    \
               ::"v"(ldsoff), "v"(gptr) : "memory")
#define ASYNC_B128_OFF48(ldsoff, gptr)                                  \
  asm volatile("global_load_async_to_lds_b128 %0, %1, off offset:48"    \
               ::"v"(ldsoff), "v"(gptr) : "memory")

__device__ __forceinline__ void wait_async0() {
  asm volatile("s_wait_asynccnt 0x0" ::: "memory");
}

// ---- WMMA fragment loaders (wave32; contiguous k order, same for A and B) ----
// A: 16x32 f16 tile from row-major src (ld in halves). Lane m=lane&15;
// lanes 0-15 carry K = koff+0..15, lanes 16-31 K = koff+16..31.
__device__ __forceinline__ v16h load_a_frag(const _Float16* base, int ld, int koff) {
  int lane = threadIdx.x & 31;
  int m  = lane & 15;
  int kb = ((lane >> 4) << 4) + koff;
  const _Float16* p = base + m * ld + kb;
  v8h lo = *(const v8h*)(p);
  v8h hi = *(const v8h*)(p + 8);
  v16h r;
#pragma unroll
  for (int i = 0; i < 8; ++i) { r[i] = lo[i]; r[8 + i] = hi[i]; }
  return r;
}

// B: 32x16 f16 tile where B[k][n] = src[n*ld + k] (src is [N][K] row-major).
// Lane n=lane&15; lanes 0-15 carry K=koff+0..15, lanes 16-31 K=koff+16..31.
__device__ __forceinline__ v16h load_b_frag(const _Float16* base, int ld, int koff) {
  int lane = threadIdx.x & 31;
  int n  = lane & 15;
  int kb = ((lane >> 4) << 4) + koff;
  return *(const v16h*)(base + n * ld + kb);
}

// ---- f32 -> f16 conversion ----
__global__ void cvt_f32_f16(const float* __restrict__ src, _Float16* __restrict__ dst,
                            int n) {
  int i = blockIdx.x * blockDim.x + threadIdx.x;
  int stride = gridDim.x * blockDim.x;
  for (; i < n; i += stride) dst[i] = (_Float16)src[i];
}

// ---- GEMM: C[M,N] = A[M,K] * W[N,K]^T, K = N = 1024 ----
// Block tile 128x64, K-step 64, async global->LDS staging.
// mode 0: store f16 head-split [B,H,S,DK] (for Q/K/V)
// mode 1: store f32 row-major [M,N] (final output projection)
__global__ __launch_bounds__(256) void gemm_xwT(const _Float16* __restrict__ A,
                                                const _Float16* __restrict__ W,
                                                _Float16* __restrict__ out16,
                                                float* __restrict__ out32,
                                                int mode) {
  __shared__ __align__(64) _Float16 At[128 * 64];
  __shared__ __align__(64) _Float16 Bt[64 * 64];
  int tid  = threadIdx.x;
  int wave = tid >> 5, lane = tid & 31;
  int m0 = blockIdx.x << 7;   // 128-row tile
  int n0 = blockIdx.y << 6;   // 64-col tile
  v8f acc0 = {}, acc1 = {}, acc2 = {}, acc3 = {};
  int arow = tid >> 1, apart = (tid & 1) << 5;  // 2 threads/row, 32 halves (64B)
  int brow = tid >> 2, bpart = (tid & 3) << 4;  // 4 threads/row, 16 halves (32B)
  unsigned at_l = lds_addr(&At[arow * 64 + apart]);
  unsigned bt_l = lds_addr(&Bt[brow * 64 + bpart]);
  const _Float16* agb = A + (size_t)(m0 + arow) * D_ + apart;
  const _Float16* wgb = W + (size_t)(n0 + brow) * D_ + bpart;
  for (int k0 = 0; k0 < D_; k0 += 64) {
    const _Float16* ag = agb + k0;
    const _Float16* wg = wgb + k0;
    ASYNC_B128(at_l, ag);
    ASYNC_B128_OFF16(at_l, ag);
    ASYNC_B128_OFF32(at_l, ag);
    ASYNC_B128_OFF48(at_l, ag);
    ASYNC_B128(bt_l, wg);
    ASYNC_B128_OFF16(bt_l, wg);
    wait_async0();         // drain this wave's ASYNCcnt
    __syncthreads();       // LDS tiles now visible to all waves
    const _Float16* Aw = &At[wave * 16 * 64];
    v16h af0 = load_a_frag(Aw, 64, 0);
    v16h af1 = load_a_frag(Aw, 64, 32);
    acc0 = wmma_ff16(af0, load_b_frag(&Bt[0 * 16 * 64], 64, 0),  acc0);
    acc0 = wmma_ff16(af1, load_b_frag(&Bt[0 * 16 * 64], 64, 32), acc0);
    acc1 = wmma_ff16(af0, load_b_frag(&Bt[1 * 16 * 64], 64, 0),  acc1);
    acc1 = wmma_ff16(af1, load_b_frag(&Bt[1 * 16 * 64], 64, 32), acc1);
    acc2 = wmma_ff16(af0, load_b_frag(&Bt[2 * 16 * 64], 64, 0),  acc2);
    acc2 = wmma_ff16(af1, load_b_frag(&Bt[2 * 16 * 64], 64, 32), acc2);
    acc3 = wmma_ff16(af0, load_b_frag(&Bt[3 * 16 * 64], 64, 0),  acc3);
    acc3 = wmma_ff16(af1, load_b_frag(&Bt[3 * 16 * 64], 64, 32), acc3);
    __syncthreads();
  }
  // epilogue: C layout -> rows m = r + 8*(lane>>4), col n = lane&15
  int mrow0 = m0 + wave * 16 + ((lane >> 4) << 3);
  int ncol  = lane & 15;
#pragma unroll
  for (int j = 0; j < 4; ++j) {
    v8f a = (j == 0) ? acc0 : (j == 1) ? acc1 : (j == 2) ? acc2 : acc3;
#pragma unroll
    for (int r = 0; r < 8; ++r) {
      int m = mrow0 + r;
      int n = n0 + j * 16 + ncol;
      if (mode == 0) {
        int bb = m >> 11, s = m & (S_ - 1);
        int hh = n >> 6,  dk = n & (DK_ - 1);
        out16[(((size_t)(bb * H_ + hh) * S_ + s) << 6) + dk] = (_Float16)a[r];
      } else {
        out32[(size_t)m * D_ + n] = a[r];
      }
    }
  }
}

// ---- RoPE on Q and K (head-split f16 buffers), one thread per rotation pair ----
__global__ __launch_bounds__(256) void rope_qk(_Float16* __restrict__ q,
                                               _Float16* __restrict__ k,
                                               const int* __restrict__ pos) {
  size_t idx = (size_t)blockIdx.x * blockDim.x + threadIdx.x;  // B*H*S*32 pairs
  int    i   = (int)(idx & (DK_ / 2 - 1));                     // pair index 0..31
  size_t row = idx >> 5;                                       // (b*H + h)*S + s
  int s  = (int)(row & (S_ - 1));
  int bb = (int)(row >> 15);                                   // row / (H_*S_)
  int p  = pos[(size_t)bb * S_ + s];
  float freq = powf(10000.0f, -(float)(2 * i) * (1.0f / (float)DK_));
  float ang  = (float)p * freq;
  float sn = sinf(ang), cs = cosf(ang);
  size_t base = (row << 6) + ((size_t)i << 1);
  float q1 = (float)q[base], q2 = (float)q[base + 1];
  q[base]     = (_Float16)(q1 * cs - q2 * sn);
  q[base + 1] = (_Float16)(q1 * sn + q2 * cs);
  float k1 = (float)k[base], k2 = (float)k[base + 1];
  k[base]     = (_Float16)(k1 * cs - k2 * sn);
  k[base + 1] = (_Float16)(k1 * sn + k2 * cs);
}

// ---- Flash attention: one wave = 16 query rows; block = 128 rows of one (b,h) ----
__global__ __launch_bounds__(256) void flash_attn(const _Float16* __restrict__ q16,
                                                  const _Float16* __restrict__ k16,
                                                  const _Float16* __restrict__ v16,
                                                  _Float16* __restrict__ attn16) {
  __shared__ __align__(64) _Float16 Kt[32 * 64];      // [key][dk]
  __shared__ __align__(64) _Float16 Vt[64 * 32];      // [dk][key] (transposed)
  __shared__ __align__(64) _Float16 Pt[8][16 * 32];   // per-wave P tile
  int tid = threadIdx.x, wave = tid >> 5, lane = tid & 31;
  int qblk = blockIdx.x, h = blockIdx.y, b = blockIdx.z;
  int q0 = (qblk << 7) + (wave << 4);
  size_t bh = (size_t)(b * H_ + h);
  const _Float16* qb  = q16 + (bh * S_ + q0) * DK_;
  const _Float16* kbp = k16 + bh * S_ * DK_;
  const _Float16* vbp = v16 + bh * S_ * DK_;
  v16h qa0 = load_a_frag(qb, DK_, 0);
  v16h qa1 = load_a_frag(qb, DK_, 32);
  v8f o0 = {}, o1 = {}, o2 = {}, o3 = {};
  float rmax[8], rsum[8];
#pragma unroll
  for (int r = 0; r < 8; ++r) { rmax[r] = -INFINITY; rsum[r] = 0.0f; }
  int mrow_off = (lane >> 4) << 3;
  int ncol = lane & 15;
  // causal: only key blocks up to this block's last query row
  int nkb = (qblk << 2) + 4;                  // blocks of 32 keys
  int kidx = tid << 3;                        // Kt: one async b128 per thread
  unsigned kt_l = lds_addr(&Kt[kidx]);
  int vky = tid & 31, vdk = (tid >> 5) << 3;  // Vt coop load (transpose on store)
  for (int kb = 0; kb < nkb; ++kb) {
    size_t koffset = (size_t)kb * 32 * DK_;
    const _Float16* kg = kbp + koffset + kidx;
    ASYNC_B128(kt_l, kg);                     // K tile straight into LDS
    v8h vv = *(const v8h*)&vbp[koffset + (size_t)vky * DK_ + vdk];
#pragma unroll
    for (int e = 0; e < 8; ++e) Vt[(vdk + e) * 32 + vky] = vv[e];
    wait_async0();
    __syncthreads();
    // S = Q * K^T : 16x32 scores, f32 accum
    v8f s0 = {}, s1 = {};
    s0 = wmma_ff16(qa0, load_b_frag(&Kt[0], DK_, 0),  s0);
    s0 = wmma_ff16(qa1, load_b_frag(&Kt[0], DK_, 32), s0);
    s1 = wmma_ff16(qa0, load_b_frag(&Kt[16 * DK_], DK_, 0),  s1);
    s1 = wmma_ff16(qa1, load_b_frag(&Kt[16 * DK_], DK_, 32), s1);
    // online softmax (rows live across a 16-lane half-wave)
    float p0[8], p1[8];
#pragma unroll
    for (int r = 0; r < 8; ++r) {
      int qg  = q0 + mrow_off + r;
      int kg0 = (kb << 5) + ncol;
      float v0 = s0[r] * 0.125f; if (kg0 > qg)      v0 = -INFINITY;
      float v1 = s1[r] * 0.125f; if (kg0 + 16 > qg) v1 = -INFINITY;
      float tmax = fmaxf(v0, v1);
      tmax = fmaxf(tmax, __shfl_xor(tmax, 1));
      tmax = fmaxf(tmax, __shfl_xor(tmax, 2));
      tmax = fmaxf(tmax, __shfl_xor(tmax, 4));
      tmax = fmaxf(tmax, __shfl_xor(tmax, 8));
      float mnew  = fmaxf(rmax[r], tmax);
      float alpha = expf(rmax[r] - mnew);
      float e0 = expf(v0 - mnew), e1 = expf(v1 - mnew);
      p0[r] = e0; p1[r] = e1;
      float tsum = e0 + e1;
      tsum += __shfl_xor(tsum, 1);
      tsum += __shfl_xor(tsum, 2);
      tsum += __shfl_xor(tsum, 4);
      tsum += __shfl_xor(tsum, 8);
      rsum[r] = rsum[r] * alpha + tsum;
      rmax[r] = mnew;
      o0[r] *= alpha; o1[r] *= alpha; o2[r] *= alpha; o3[r] *= alpha;
    }
    // round-trip P through per-wave LDS to re-lay as A fragment (same-wave DS in-order)
    _Float16* Pw = &Pt[wave][0];
#pragma unroll
    for (int r = 0; r < 8; ++r) {
      int m = mrow_off + r;
      Pw[m * 32 + ncol]      = (_Float16)p0[r];
      Pw[m * 32 + 16 + ncol] = (_Float16)p1[r];
    }
    v16h pa = load_a_frag(Pw, 32, 0);
    // O += P(16x32) * V(32x64); V stored transposed so B-frag loads are contiguous
    o0 = wmma_ff16(pa, load_b_frag(&Vt[0 * 16 * 32], 32, 0), o0);
    o1 = wmma_ff16(pa, load_b_frag(&Vt[1 * 16 * 32], 32, 0), o1);
    o2 = wmma_ff16(pa, load_b_frag(&Vt[2 * 16 * 32], 32, 0), o2);
    o3 = wmma_ff16(pa, load_b_frag(&Vt[3 * 16 * 32], 32, 0), o3);
    __syncthreads();
  }
  // normalize and write attention output as f16 in [B,S,D] (e = h*64 + dk)
  _Float16* outp = attn16 + (size_t)b * S_ * D_;
#pragma unroll
  for (int r = 0; r < 8; ++r) {
    int qg = q0 + mrow_off + r;
    float inv = 1.0f / rsum[r];
    size_t rowbase = (size_t)qg * D_ + (size_t)h * DK_;
    outp[rowbase + 0  + ncol] = (_Float16)(o0[r] * inv);
    outp[rowbase + 16 + ncol] = (_Float16)(o1[r] * inv);
    outp[rowbase + 32 + ncol] = (_Float16)(o2[r] * inv);
    outp[rowbase + 48 + ncol] = (_Float16)(o3[r] * inv);
  }
}

extern "C" void kernel_launch(void* const* d_in, const int* in_sizes, int n_in,
                              void* d_out, int out_size, void* d_ws, size_t ws_size,
                              hipStream_t stream) {
  (void)in_sizes; (void)n_in; (void)out_size;
  const float* x  = (const float*)d_in[0];
  const float* Wq = (const float*)d_in[1];
  const float* Wk = (const float*)d_in[2];
  const float* Wv = (const float*)d_in[3];
  const float* Wo = (const float*)d_in[4];
  const int*  pos = (const int*)d_in[5];

  char* p = (char*)d_ws;
  auto carve = [&](size_t halves) -> _Float16* {
    _Float16* r = (_Float16*)p;
    p += ((halves * sizeof(_Float16)) + 255) & ~(size_t)255;
    return r;
  };
  const size_t ND = (size_t)B_ * S_ * D_;  // 4 Mi elems
  const size_t NW = (size_t)D_ * D_;       // 1 Mi elems
  _Float16* x16  = carve(ND);
  _Float16* wq16 = carve(NW);
  _Float16* wk16 = carve(NW);
  _Float16* wv16 = carve(NW);
  _Float16* wo16 = carve(NW);
  _Float16* q16  = carve(ND);
  _Float16* k16  = carve(ND);
  _Float16* v16  = carve(ND);
  _Float16* a16  = carve(ND);
  if ((size_t)(p - (char*)d_ws) > ws_size) return;  // insufficient scratch

  cvt_f32_f16<<<2048, 256, 0, stream>>>(x,  x16,  (int)ND);
  cvt_f32_f16<<<1024, 256, 0, stream>>>(Wq, wq16, (int)NW);
  cvt_f32_f16<<<1024, 256, 0, stream>>>(Wk, wk16, (int)NW);
  cvt_f32_f16<<<1024, 256, 0, stream>>>(Wv, wv16, (int)NW);
  cvt_f32_f16<<<1024, 256, 0, stream>>>(Wo, wo16, (int)NW);

  dim3 ggrid(B_ * S_ / 128, D_ / 64);  // (32, 16)
  gemm_xwT<<<ggrid, 256, 0, stream>>>(x16, wq16, q16, nullptr, 0);
  gemm_xwT<<<ggrid, 256, 0, stream>>>(x16, wk16, k16, nullptr, 0);
  gemm_xwT<<<ggrid, 256, 0, stream>>>(x16, wv16, v16, nullptr, 0);

  rope_qk<<<(B_ * H_ * S_ * (DK_ / 2)) / 256, 256, 0, stream>>>(q16, k16, pos);

  flash_attn<<<dim3(S_ / 128, H_, B_), 256, 0, stream>>>(q16, k16, v16, a16);

  gemm_xwT<<<ggrid, 256, 0, stream>>>(a16, wo16, nullptr, (float*)d_out, 1);
}
